// ContextAttentionMaskLuong_80805514707578
// MI455X (gfx1250) — compile-verified, matrix operand order checked by reading, not run on hardware
//
#include <hip/hip_runtime.h>
#include <hip/hip_bf16.h>
#include <hip/hip_fp16.h>

// ContextAttentionMaskLuong, MI455X (gfx1250, wave32).
//
// Math refactor: eij[b,s] = query[b,:] . (x[b,s,:] @ W) = x[b,s,:] . (W @ query[b])
// => never materialize keys (saves a 68.7 GFLOP GEMM + 128 MiB round trip).
// Pipeline:
//   K1: f32->f16 convert of W (used as-is: Bt[n=d][k=e] = W[d,e]) and c
//   K2: tiled LDS transpose+convert of Wc -> WcT[n=d][k=cc] (f16)
//   K3: WMMA GEMM  query = tanh(c @ Wc + b)            [16x1024, K=1024] -> f16
//   K4: WMMA GEMM  qtilde = scale * (query @ W^T)      [16x1024, K=1024] -> f32
//   K5: eij[b,s] = x[b,s,:] . qtilde[b,:]   (one wave32 per row, float4 loads)
//   K6: masked softmax over S per batch (mask applied after exp, per reference)
//   K7: partial weighted sums over 16 s-chunks (deterministic, no atomics)
//   K8: reduce partials -> out[b,d]

typedef __attribute__((ext_vector_type(16))) _Float16 v16h;
typedef __attribute__((ext_vector_type(8)))  _Float16 v8h;
typedef __attribute__((ext_vector_type(8)))  float    v8f;

#define B_   16
#define S_   2048
#define D_   1024
#define EPS_ 1e-7f

// ---------------- K1: convert W and c to f16 ----------------
__global__ void convert_kernel(const float* __restrict__ W,
                               const float* __restrict__ c,
                               _Float16* __restrict__ W_h,
                               _Float16* __restrict__ c_h) {
    int idx = blockIdx.x * blockDim.x + threadIdx.x;   // 0 .. 1024*1024-1
    W_h[idx] = (_Float16)W[idx];
    if (idx < B_ * D_) c_h[idx] = (_Float16)c[idx];
}

// ---------------- K2: transpose+convert Wc -> WcT (f16) ----------------
__global__ void transpose_conv_kernel(const float* __restrict__ Wc,
                                      _Float16* __restrict__ WcT) {
    __shared__ _Float16 tile[32][33];
    int x  = blockIdx.x * 32 + threadIdx.x;   // d (input col)
    int y0 = blockIdx.y * 32;                 // cc (input row base)
    for (int i = threadIdx.y; i < 32; i += 8)
        tile[i][threadIdx.x] = (_Float16)Wc[(y0 + i) * D_ + x];
    __syncthreads();
    int xo  = blockIdx.y * 32 + threadIdx.x;  // cc (output col)
    int yo0 = blockIdx.x * 32;                // d  (output row base)
    for (int i = threadIdx.y; i < 32; i += 8)
        WcT[(yo0 + i) * D_ + xo] = tile[threadIdx.x][i];
}

// Load a 16x32 f16 A tile (row r = lane&15) / 32x16 B tile (col n = lane&15)
// into the ISA VGPR layout: lanes 0-15 hold K k0..k0+7 (v0-3) and k0+16..23
// (v4-7); lanes 16-31 hold k0+8..15 and k0+24..31.
__device__ inline v16h load_tile_rowmajor(const _Float16* __restrict__ base,
                                          int row, int k0, int grp) {
    v8h lo = *(const v8h*)(base + row * D_ + k0 + grp * 8);
    v8h hi = *(const v8h*)(base + row * D_ + k0 + 16 + grp * 8);
    v16h v;
#pragma unroll
    for (int i = 0; i < 8; ++i) { v[i] = lo[i]; v[i + 8] = hi[i]; }
    return v;
}

// ---------------- K3: query = tanh(c @ Wc + b), f16 out ----------------
// A = c_h [16][1024] row-major, Bt = WcT [n=1024][k=1024] row-major.
__global__ void gemm_tanh_kernel(const _Float16* __restrict__ A,
                                 const _Float16* __restrict__ Bt,
                                 const float* __restrict__ bias,
                                 _Float16* __restrict__ out_h) {
    int wave  = threadIdx.x >> 5;
    int lane  = threadIdx.x & 31;
    int ntile = blockIdx.x * (blockDim.x >> 5) + wave;   // 0..63
    int r     = lane & 15;
    int grp   = lane >> 4;
    int ncol  = ntile * 16 + r;
    v8f acc = {};
    for (int k0 = 0; k0 < D_; k0 += 32) {
        v16h av = load_tile_rowmajor(A,  r,    k0, grp);
        v16h bv = load_tile_rowmajor(Bt, ncol, k0, grp);
        acc = __builtin_amdgcn_wmma_f32_16x16x32_f16(false, av, false, bv,
                                                     (short)0, acc, false, false);
    }
    int n     = ntile * 16 + (lane & 15);
    int mbase = (lane >> 4) * 8;
    float bn  = bias[n];
#pragma unroll
    for (int j = 0; j < 8; ++j) {
        int m = mbase + j;
        out_h[m * D_ + n] = (_Float16)tanhf(acc[j] + bn);
    }
}

// ---------------- K4: qtilde = scale * (query @ W^T), f32 out ----------------
// A = query_h [16][1024], Bt = W_h (W row-major already is Bt[n=d][k=e]).
__global__ void gemm_scale_kernel(const _Float16* __restrict__ A,
                                  const _Float16* __restrict__ Bt,
                                  const float* __restrict__ scale,
                                  float* __restrict__ qtilde) {
    int wave  = threadIdx.x >> 5;
    int lane  = threadIdx.x & 31;
    int ntile = blockIdx.x * (blockDim.x >> 5) + wave;
    int r     = lane & 15;
    int grp   = lane >> 4;
    int ncol  = ntile * 16 + r;
    v8f acc = {};
    for (int k0 = 0; k0 < D_; k0 += 32) {
        v16h av = load_tile_rowmajor(A,  r,    k0, grp);
        v16h bv = load_tile_rowmajor(Bt, ncol, k0, grp);
        acc = __builtin_amdgcn_wmma_f32_16x16x32_f16(false, av, false, bv,
                                                     (short)0, acc, false, false);
    }
    float sc  = scale[0];
    int n     = ntile * 16 + (lane & 15);
    int mbase = (lane >> 4) * 8;
#pragma unroll
    for (int j = 0; j < 8; ++j) {
        int m = mbase + j;
        qtilde[m * D_ + n] = sc * acc[j];
    }
}

// ---------------- K5: eij[b,s] = x[b,s,:] . qtilde[b,:] ----------------
// One wave32 per (b,s) row; 8 float4 loads per lane; shfl_xor reduction.
__global__ void scores_kernel(const float* __restrict__ x,
                              const float* __restrict__ qtilde,
                              float* __restrict__ eij) {
    int wave = threadIdx.x >> 5;
    int lane = threadIdx.x & 31;
    int row  = blockIdx.x * (blockDim.x >> 5) + wave;    // b*S + s
    int b    = row >> 11;                                // S = 2048
    const float4* xr = (const float4*)(x + (size_t)row * D_);
    const float4* qr = (const float4*)(qtilde + b * D_);
    float acc = 0.f;
#pragma unroll
    for (int i = 0; i < 8; ++i) {
        float4 xv = xr[i * 32 + lane];
        float4 qv = qr[i * 32 + lane];
        acc += xv.x * qv.x + xv.y * qv.y + xv.z * qv.z + xv.w * qv.w;
    }
#pragma unroll
    for (int m = 16; m >= 1; m >>= 1) acc += __shfl_xor(acc, m, 32);
    if (lane == 0) eij[row] = acc;
}

// ---------------- K6: masked softmax over S (mask applied after exp) -------
__global__ void softmax_kernel(const float* __restrict__ eij,
                               const int* __restrict__ mask,
                               float* __restrict__ a) {
    __shared__ float red[256];
    int b = blockIdx.x;
    int t = threadIdx.x;
    float vals[8];
    float mx = -INFINITY;
#pragma unroll
    for (int i = 0; i < 8; ++i) {
        vals[i] = eij[b * S_ + i * 256 + t];
        mx = fmaxf(mx, vals[i]);
    }
    red[t] = mx; __syncthreads();
    for (int off = 128; off >= 1; off >>= 1) {
        if (t < off) red[t] = fmaxf(red[t], red[t + off]);
        __syncthreads();
    }
    mx = red[0]; __syncthreads();
    float ex[8];
    float sum = 0.f;
#pragma unroll
    for (int i = 0; i < 8; ++i) {
        int s = i * 256 + t;
        float e = expf(vals[i] - mx) * (float)mask[b * S_ + s];
        ex[i] = e; sum += e;
    }
    red[t] = sum; __syncthreads();
    for (int off = 128; off >= 1; off >>= 1) {
        if (t < off) red[t] += red[t + off];
        __syncthreads();
    }
    float inv = 1.f / (red[0] + EPS_);
#pragma unroll
    for (int i = 0; i < 8; ++i) a[b * S_ + i * 256 + t] = ex[i] * inv;
}

// ---------------- K7: partial weighted sums over 16 s-chunks ----------------
// block = (b, schunk); 256 threads cover D with 4 accumulators each.
__global__ void wsum_partial_kernel(const float* __restrict__ x,
                                    const float* __restrict__ a,
                                    float* __restrict__ part) {   // [B][16][D]
    int b  = blockIdx.x >> 4;
    int sc = blockIdx.x & 15;
    int t  = threadIdx.x;
    int s0 = sc * 128;
    const float* xb = x + ((size_t)b * S_ + s0) * D_;
    const float* ab = a + b * S_ + s0;
    float a0 = 0.f, a1 = 0.f, a2 = 0.f, a3 = 0.f;
    for (int s = 0; s < 128; ++s) {
        float as = ab[s];
        const float* xr = xb + (size_t)s * D_;
        a0 += as * xr[t];
        a1 += as * xr[t + 256];
        a2 += as * xr[t + 512];
        a3 += as * xr[t + 768];
    }
    float* p = part + (b * 16 + sc) * D_;
    p[t] = a0; p[t + 256] = a1; p[t + 512] = a2; p[t + 768] = a3;
}

// ---------------- K8: reduce partials -> out[b,d] ----------------
__global__ void wsum_reduce_kernel(const float* __restrict__ part,
                                   float* __restrict__ out) {
    int idx = blockIdx.x * blockDim.x + threadIdx.x;   // 0..16383
    int b = idx >> 10, d = idx & 1023;
    float s = 0.f;
#pragma unroll
    for (int c = 0; c < 16; ++c) s += part[(b * 16 + c) * D_ + d];
    out[idx] = s;
}

extern "C" void kernel_launch(void* const* d_in, const int* in_sizes, int n_in,
                              void* d_out, int out_size, void* d_ws, size_t ws_size,
                              hipStream_t stream) {
    (void)in_sizes; (void)n_in; (void)out_size; (void)ws_size;
    const float* x     = (const float*)d_in[0];   // [16,2048,1024]
    const int*   mask  = (const int*)  d_in[1];   // [16,2048]
    const float* c     = (const float*)d_in[2];   // [16,1024]
    const float* W     = (const float*)d_in[3];   // [1024,1024]
    const float* Wc    = (const float*)d_in[4];   // [1024,1024]
    const float* bias  = (const float*)d_in[5];   // [1024]
    const float* scale = (const float*)d_in[6];   // [1]
    float* out = (float*)d_out;                   // [16,1024]

    // Workspace carve-up (256B aligned offsets).
    char* ws = (char*)d_ws;
    size_t off = 0;
    auto carve = [&](size_t bytes) { char* p = ws + off; off += (bytes + 255) & ~size_t(255); return p; };
    _Float16* c_h    = (_Float16*)carve((size_t)B_ * D_ * 2);       //  32 KB
    _Float16* W_h    = (_Float16*)carve((size_t)D_ * D_ * 2);       //   2 MB
    _Float16* WcT_h  = (_Float16*)carve((size_t)D_ * D_ * 2);       //   2 MB
    _Float16* q_h    = (_Float16*)carve((size_t)B_ * D_ * 2);       //  32 KB
    float*    qtilde = (float*)   carve((size_t)B_ * D_ * 4);       //  64 KB
    float*    eij    = (float*)   carve((size_t)B_ * S_ * 4);       // 128 KB
    float*    attn   = (float*)   carve((size_t)B_ * S_ * 4);       // 128 KB
    float*    part   = (float*)   carve((size_t)B_ * 16 * D_ * 4);  //   1 MB

    // K1: f32->f16 (W as-is, c)
    convert_kernel<<<(D_ * D_) / 256, 256, 0, stream>>>(W, c, W_h, c_h);
    // K2: Wc -> WcT (f16, transposed)
    transpose_conv_kernel<<<dim3(D_ / 32, D_ / 32), dim3(32, 8), 0, stream>>>(Wc, WcT_h);
    // K3: query = tanh(c @ Wc + b)   (64 N-tiles: 16 blocks x 4 waves)
    gemm_tanh_kernel<<<16, 128, 0, stream>>>(c_h, WcT_h, bias, q_h);
    // K4: qtilde = scale * (query @ W^T)
    gemm_scale_kernel<<<16, 128, 0, stream>>>(q_h, W_h, scale, qtilde);
    // K5: eij (one wave per row; 32768 rows / 8 waves per block)
    scores_kernel<<<(B_ * S_) / 8, 256, 0, stream>>>(x, qtilde, eij);
    // K6: masked softmax per batch
    softmax_kernel<<<B_, 256, 0, stream>>>(eij, mask, attn);
    // K7: deterministic partial weighted sums (16 s-chunks per batch)
    wsum_partial_kernel<<<B_ * 16, 256, 0, stream>>>(x, attn, part);
    // K8: reduce partials into out
    wsum_reduce_kernel<<<(B_ * D_) / 256, 256, 0, stream>>>(part, out);
}
